// CausalMaskAttention_83803401879679
// MI455X (gfx1250) — matrix operand order repaired and checked by reading, not compiled, not run on hardware
//
#include <hip/hip_runtime.h>
#include <hip/hip_bf16.h>

// CDNA5 / gfx1250 causal MHA: QKV proj (WMMA bf16, software-pipelined) ->
// transposed flash attention (WMMA bf16, exp2-domain online softmax,
// mask-free main loop, 32 q-cols/wave) -> output proj (WMMA bf16).

#define DEV __device__ __forceinline__

typedef __attribute__((ext_vector_type(16))) __bf16 v16bf;
typedef __attribute__((ext_vector_type(8)))  __bf16 bf16x8;
typedef __attribute__((ext_vector_type(8)))  float  v8f;
typedef __attribute__((ext_vector_type(4)))  float  f32x4;

union FragU { v16bf v; bf16x8 h[2]; unsigned u[8]; };

DEV v8f wmma_bf16(v16bf a, v16bf b, v8f c) {
  // D = A(16x32 bf16) x B(32x16 bf16) + C(16x16 f32)
  return __builtin_amdgcn_wmma_f32_16x16x32_bf16(false, a, false, b,
                                                 (short)0, c, false, false);
}

DEV unsigned short bf16_bits(float f) {
  return __builtin_bit_cast(unsigned short, (__bf16)f);
}

constexpr int BATCH = 2, HEADS = 16, SEQ = 2048, DHEAD = 64;
constexpr int DMODEL = 1024;
constexpr int MTOT = BATCH * SEQ;  // 4096
constexpr int LDSPAD = 40;         // 32 + 8 bf16 pad: fragment gathers spread over banks

// ---------------------------------------------------------------------------
// Register staging for one 128x32 A tile + 32x128 W tile (pipelined prefetch)
// ---------------------------------------------------------------------------
template <typename AT>
struct Stage {
  f32x4  a0, a1, a2, a3;   // float A path
  bf16x8 ab0, ab1;         // bf16 A path
  f32x4  w00, w01, w10, w11;

  DEV void fetch(const AT* __restrict__ A, const float* __restrict__ W,
                 int m0, int n0, int k0, int arow, int acol, int kp, int ncg) {
    if constexpr (__is_same(AT, float)) {
      const float* src = A + (size_t)(m0 + arow) * DMODEL + k0 + acol;
      a0 = *(const f32x4*)(src);
      a1 = *(const f32x4*)(src + 4);
      a2 = *(const f32x4*)(src + 8);
      a3 = *(const f32x4*)(src + 12);
    } else {
      const AT* src = A + (size_t)(m0 + arow) * DMODEL + k0 + acol;
      ab0 = *(const bf16x8*)(src);
      ab1 = *(const bf16x8*)(src + 8);
    }
    const float* r0 = W + (size_t)(k0 + 2 * kp) * DMODEL + n0 + ncg;
    w00 = *(const f32x4*)(r0);
    w01 = *(const f32x4*)(r0 + 4);
    w10 = *(const f32x4*)(r0 + DMODEL);
    w11 = *(const f32x4*)(r0 + DMODEL + 4);
  }

  DEV void commit(__bf16* As, __bf16* Bs, int arow, int acol, int kp, int ncg) {
    __bf16* dst = &As[arow * LDSPAD + acol];
    if constexpr (__is_same(AT, float)) {
      bf16x8 h0, h1;
#pragma unroll
      for (int i = 0; i < 4; ++i) {
        h0[i] = (__bf16)a0[i]; h0[i + 4] = (__bf16)a1[i];
        h1[i] = (__bf16)a2[i]; h1[i + 4] = (__bf16)a3[i];
      }
      *(bf16x8*)(dst)     = h0;
      *(bf16x8*)(dst + 8) = h1;
    } else {
      *(bf16x8*)(dst)     = ab0;
      *(bf16x8*)(dst + 8) = ab1;
    }
    // W transposed into Bs[n][k], (k,k+1) bf16 pairs packed per dword
#pragma unroll
    for (int i = 0; i < 8; ++i) {
      float lo  = (i < 4) ? w00[i] : w01[i - 4];
      float hiv = (i < 4) ? w10[i] : w11[i - 4];
      unsigned pk = (unsigned)bf16_bits(lo) | ((unsigned)bf16_bits(hiv) << 16);
      *(unsigned*)&Bs[(ncg + i) * LDSPAD + 2 * kp] = pk;
    }
  }
};

// ---------------------------------------------------------------------------
// Tiled GEMM: C[M=4096, N=1024] = A[M,1024] * W[1024,N] + bias
// Block tile 128x128, 8 waves (2x4), wave tile 64x32 -> 4x2 WMMA accs.
// OUT_MODE 0: bf16 [B,H,S,d] (Q,K)   1: bf16 [B,H,d,S] (V^T)   2: f32 [M,N]
// ---------------------------------------------------------------------------
template <typename AT, int OUT_MODE>
__global__ __launch_bounds__(256) void gemm_wmma_kernel(
    const AT* __restrict__ A, const float* __restrict__ W,
    const float* __restrict__ bias, void* __restrict__ Out) {
  __shared__ __align__(16) __bf16 As[128 * LDSPAD];  // [m][k]
  __shared__ __align__(16) __bf16 Bs[128 * LDSPAD];  // [n][k]

  const int tid  = threadIdx.x;
  const int lane = tid & 31;
  const int wave = tid >> 5;
  const int wm = wave >> 2, wn = wave & 3;
  const int l16 = lane & 15, hi = lane >> 4;

  const int n0 = blockIdx.x * 128;
  const int m0 = blockIdx.y * 128;

  const int arow = tid >> 1;         // 0..127
  const int acol = (tid & 1) * 16;   // 0 / 16
  const int kp   = tid >> 4;         // 0..15 -> k pair (2kp, 2kp+1)
  const int ncg  = (tid & 15) * 8;   // 8-wide column group

  v8f acc[4][2] = {};
  Stage<AT> stg;
  stg.fetch(A, W, m0, n0, 0, arow, acol, kp, ncg);

  for (int k0 = 0; k0 < DMODEL; k0 += 32) {
    stg.commit(As, Bs, arow, acol, kp, ncg);
    __syncthreads();
    if (k0 + 32 < DMODEL)  // prefetch next tile: overlaps ds-loads + WMMAs
      stg.fetch(A, W, m0, n0, k0 + 32, arow, acol, kp, ncg);

    FragU af[4], bfr[2];
#pragma unroll
    for (int i = 0; i < 4; ++i) {
      const __bf16* p = &As[(wm * 64 + i * 16 + l16) * LDSPAD + hi * 8];
      af[i].h[0] = *(const bf16x8*)(p);        // K kh+0..7
      af[i].h[1] = *(const bf16x8*)(p + 16);   // K kh+16..23
    }
#pragma unroll
    for (int j = 0; j < 2; ++j) {
      const __bf16* p = &Bs[(wn * 32 + j * 16 + l16) * LDSPAD + hi * 8];
      bfr[j].h[0] = *(const bf16x8*)(p);       // K kh+0..7
      bfr[j].h[1] = *(const bf16x8*)(p + 16);  // K kh+16..23
    }
#pragma unroll
    for (int i = 0; i < 4; ++i)
#pragma unroll
      for (int j = 0; j < 2; ++j)
        acc[i][j] = wmma_bf16(af[i].v, bfr[j].v, acc[i][j]);
    __syncthreads();
  }

  // ---- epilogue: bias + store per OUT_MODE ----
#pragma unroll
  for (int i = 0; i < 4; ++i) {
    const int mbase = m0 + wm * 64 + i * 16 + hi * 8;
#pragma unroll
    for (int j = 0; j < 2; ++j) {
      const int ncol = n0 + wn * 32 + j * 16 + l16;
      const float bv = bias[ncol];
      if constexpr (OUT_MODE == 2) {
        float* O = (float*)Out;
#pragma unroll
        for (int r = 0; r < 8; ++r)
          O[(size_t)(mbase + r) * DMODEL + ncol] = acc[i][j][r] + bv;
      } else if constexpr (OUT_MODE == 0) {  // bf16 [B,H,S,d]
        __bf16* O = (__bf16*)Out;
        const int h = ncol >> 6, dd = ncol & 63;
#pragma unroll
        for (int r = 0; r < 8; ++r) {
          const int m = mbase + r;
          const int bb = m >> 11, s = m & (SEQ - 1);
          O[(((size_t)bb * HEADS + h) * SEQ + s) * DHEAD + dd] =
              (__bf16)(acc[i][j][r] + bv);
        }
      } else {  // OUT_MODE == 1 : V^T bf16 [B,H,d,S] — 8 rows contiguous in S
        __bf16* O = (__bf16*)Out;
        const int h = ncol >> 6, dd = ncol & 63;
        const int bb = mbase >> 11, s = mbase & (SEQ - 1);
        bf16x8 pk;
#pragma unroll
        for (int r = 0; r < 8; ++r) pk[r] = (__bf16)(acc[i][j][r] + bv);
        *(bf16x8*)&O[(((size_t)bb * HEADS + h) * DHEAD + dd) * SEQ + s] = pk;
      }
    }
  }
}

// ---------------------------------------------------------------------------
// Transposed flash attention, 32 q columns per wave (two 16-col groups
// sharing K/V fragments).  S^T = K x Q^T, O^T = V^T x P^T.
// Softmax runs in the exp2 domain (v_exp_f32 is base-2 natively); causal
// masking only in the single diagonal step (k0 == q0).
// ---------------------------------------------------------------------------
template <bool MASKED>
DEV void attn_step(int k0, const __bf16* __restrict__ Kh,
                   const __bf16* __restrict__ Vh,
                   const FragU (&qf)[2][2], v8f (&accT)[2][4],
                   float (&mrun)[2], float (&lrun)[2],
                   int qcol0, int l16, int hi) {
  constexpr float CSCALE = 0.1803368801111244f;  // (1/sqrt(64)) * log2(e)

  // K A-fragments: 2 k-tiles x 2 depth chunks (shared by both q-groups)
  FragU kf[2][2];
  {
    const __bf16* kp = Kh + (size_t)(k0 + l16) * DHEAD + hi * 16;
#pragma unroll
    for (int t = 0; t < 2; ++t) {
      kf[t][0].h[0] = *(const bf16x8*)(kp);
      kf[t][0].h[1] = *(const bf16x8*)(kp + 8);
      kf[t][1].h[0] = *(const bf16x8*)(kp + 32);
      kf[t][1].h[1] = *(const bf16x8*)(kp + 40);
      kp += 16 * DHEAD;
    }
  }

  FragU pf[2];
#pragma unroll
  for (int g = 0; g < 2; ++g) {
    v8f st0 = {}, st1 = {};
    st0 = wmma_bf16(kf[0][0].v, qf[g][0].v, st0);
    st0 = wmma_bf16(kf[0][1].v, qf[g][1].v, st0);
    st1 = wmma_bf16(kf[1][0].v, qf[g][0].v, st1);
    st1 = wmma_bf16(kf[1][1].v, qf[g][1].v, st1);

    const int qcol = qcol0 + g * 16;
    const int kb0 = k0 + hi * 8, kb1 = kb0 + 16;
    float p[16];
#pragma unroll
    for (int r = 0; r < 8; ++r) {
      float t0 = st0[r] * CSCALE;
      float t1 = st1[r] * CSCALE;
      if constexpr (MASKED) {
        t0 = (kb0 + r <= qcol) ? t0 : -1e30f;
        t1 = (kb1 + r <= qcol) ? t1 : -1e30f;
      }
      p[r] = t0;
      p[r + 8] = t1;
    }

    float vmax = p[0];
#pragma unroll
    for (int i = 1; i < 16; ++i) vmax = fmaxf(vmax, p[i]);
    vmax = fmaxf(vmax, __shfl_xor(vmax, 16, 32));
    const float mnew = fmaxf(mrun[g], vmax);
    const float alpha = exp2f(mrun[g] - mnew);
    mrun[g] = mnew;
    float rs = 0.0f;
#pragma unroll
    for (int i = 0; i < 16; ++i) { p[i] = exp2f(p[i] - mnew); rs += p[i]; }
    rs += __shfl_xor(rs, 16, 32);
    lrun[g] = lrun[g] * alpha + rs;
#pragma unroll
    for (int j = 0; j < 4; ++j)
#pragma unroll
      for (int r = 0; r < 8; ++r) accT[g][j][r] *= alpha;

    // P^T B-fragment: pack bf16 pairs, half-swap with lane^16.
    // hi=0 holds k0+0..7 / k0+16..23; hi=1 holds k0+8..15 / k0+24..31.
    unsigned ownlo[4], ownhi[4], rcv[4];
#pragma unroll
    for (int i = 0; i < 4; ++i) {
      ownlo[i] = (unsigned)bf16_bits(p[2 * i]) |
                 ((unsigned)bf16_bits(p[2 * i + 1]) << 16);
      ownhi[i] = (unsigned)bf16_bits(p[8 + 2 * i]) |
                 ((unsigned)bf16_bits(p[8 + 2 * i + 1]) << 16);
    }
#pragma unroll
    for (int i = 0; i < 4; ++i) {
      const unsigned snd = hi ? ownlo[i] : ownhi[i];
      rcv[i] = (unsigned)__shfl_xor((int)snd, 16, 32);
    }
#pragma unroll
    for (int i = 0; i < 4; ++i) {
      pf[g].u[i]     = hi ? rcv[i]   : ownlo[i];
      pf[g].u[i + 4] = hi ? ownhi[i] : rcv[i];
    }
  }

  // V A-fragments shared by both q-groups -> 8 PV WMMAs
#pragma unroll
  for (int j = 0; j < 4; ++j) {
    const __bf16* vp = Vh + (size_t)(j * 16 + l16) * SEQ + k0 + hi * 16;
    FragU vf;
    vf.h[0] = *(const bf16x8*)(vp);
    vf.h[1] = *(const bf16x8*)(vp + 8);
    accT[0][j] = wmma_bf16(vf.v, pf[0].v, accT[0][j]);
    accT[1][j] = wmma_bf16(vf.v, pf[1].v, accT[1][j]);
  }
}

__global__ __launch_bounds__(128) void flash_attn_kernel(
    const __bf16* __restrict__ Q, const __bf16* __restrict__ K,
    const __bf16* __restrict__ Vt, __bf16* __restrict__ O) {
  const int tid  = threadIdx.x;
  const int lane = tid & 31;
  const int wave = tid >> 5;                 // 0..3
  const int l16 = lane & 15, hi = lane >> 4;

  const int qb = blockIdx.x & 15;            // 16 q-blocks of 128
  const int bh = blockIdx.x >> 4;            // 0..31
  const int b = bh >> 4, h = bh & 15;

  const int q0 = qb * 128 + wave * 32;       // 32 q columns per wave
  const int qcol0 = q0 + l16;
  const __bf16* Kh = K + (size_t)bh * SEQ * DHEAD;
  const __bf16* Vh = Vt + (size_t)bh * DHEAD * SEQ;

  // Q^T B-fragments, resident: lane = column q, contraction depth hi*16+0..15
  FragU qf[2][2];
#pragma unroll
  for (int g = 0; g < 2; ++g) {
    const __bf16* qp = Q + (size_t)bh * SEQ * DHEAD +
                       (size_t)(q0 + g * 16 + l16) * DHEAD + hi * 16;
    qf[g][0].h[0] = *(const bf16x8*)(qp);
    qf[g][0].h[1] = *(const bf16x8*)(qp + 8);
    qf[g][1].h[0] = *(const bf16x8*)(qp + 32);
    qf[g][1].h[1] = *(const bf16x8*)(qp + 40);
  }

  v8f accT[2][4] = {};
  float mrun[2] = {-1e30f, -1e30f};
  float lrun[2] = {0.0f, 0.0f};

  // mask-free main loop: every k-block strictly below the q-tile's diagonal
  for (int k0 = 0; k0 < q0; k0 += 32)
    attn_step<false>(k0, Kh, Vh, qf, accT, mrun, lrun, qcol0, l16, hi);
  // single diagonal step with causal masking (covers k = q0 .. q0+31)
  attn_step<true>(q0, Kh, Vh, qf, accT, mrun, lrun, qcol0, l16, hi);

  // ---- normalize; O^T layout gives contiguous d per lane: b128 stores ----
#pragma unroll
  for (int g = 0; g < 2; ++g) {
    const float inv = 1.0f / lrun[g];
    const int qcol = qcol0 + g * 16;
#pragma unroll
    for (int j = 0; j < 4; ++j) {
      bf16x8 pk;
#pragma unroll
      for (int r = 0; r < 8; ++r) pk[r] = (__bf16)(accT[g][j][r] * inv);
      __bf16* op = O + ((size_t)b * SEQ + qcol) * DMODEL + h * DHEAD +
                   j * 16 + hi * 8;
      *(bf16x8*)op = pk;
    }
  }
}

// ---------------------------------------------------------------------------
extern "C" void kernel_launch(void* const* d_in, const int* in_sizes, int n_in,
                              void* d_out, int out_size, void* d_ws, size_t ws_size,
                              hipStream_t stream) {
  (void)in_sizes; (void)n_in; (void)out_size; (void)ws_size;
  const float* xk = (const float*)d_in[0];
  const float* xq = (const float*)d_in[1];
  const float* xv = (const float*)d_in[2];
  const float* wq = (const float*)d_in[3];
  const float* bq = (const float*)d_in[4];
  const float* wk = (const float*)d_in[5];
  const float* bk = (const float*)d_in[6];
  const float* wv = (const float*)d_in[7];
  const float* bv = (const float*)d_in[8];
  const float* wo = (const float*)d_in[9];
  const float* bo = (const float*)d_in[10];

  const size_t NEL = (size_t)MTOT * DMODEL;  // 4,194,304 elems per buffer
  __bf16* Qb  = (__bf16*)d_ws;
  __bf16* Kb  = Qb + NEL;
  __bf16* Vtb = Kb + NEL;
  __bf16* Ob  = Vtb + NEL;   // 4 x 8MB = 32MB workspace

  dim3 gg(DMODEL / 128, MTOT / 128);
  gemm_wmma_kernel<float, 0><<<gg, 256, 0, stream>>>(xq, wq, bq, (void*)Qb);
  gemm_wmma_kernel<float, 0><<<gg, 256, 0, stream>>>(xk, wk, bk, (void*)Kb);
  gemm_wmma_kernel<float, 1><<<gg, 256, 0, stream>>>(xv, wv, bv, (void*)Vtb);

  flash_attn_kernel<<<dim3(BATCH * HEADS * (SEQ / 128)), 128, 0, stream>>>(
      Qb, Kb, Vtb, Ob);

  gemm_wmma_kernel<__bf16, 2><<<gg, 256, 0, stream>>>(Ob, wo, bo, d_out);
}
